// SPAIRPointFeatureNetwork_64269890617419
// MI455X (gfx1250) — matrix operand order repaired and checked by reading, not compiled, not run on hardware
//
#include <hip/hip_runtime.h>
#include <hip/hip_bf16.h>
#include <math.h>

typedef __attribute__((ext_vector_type(16))) _Float16 v16h;
typedef __attribute__((ext_vector_type(8)))  float    v8f;

#define NPTS 65536
#define KNBR 64

__device__ __forceinline__ float celu1(float x) {
    // celu(alpha=1) = max(x,0) + exp(min(x,0)) - 1   (single v_exp_f32 TRANS op)
    return fmaxf(x, 0.0f) + __expf(fminf(x, 0.0f)) - 1.0f;
}

// ---------------------------------------------------------------------------
// WMMA operand builders. G (lane group = lane>>4) is a template parameter so
// every k is compile-time: predicates fold away and feature fetches become
// unconditional contiguous loads the compiler can merge into b64/b128.
//
// A layout (16x32 f16): lane l -> row M=l%16, half h -> K=(h<8?h:h+8)+G*8
// B layout (32x16 f16): lane l -> col N=l%16, half h -> K=(h<8?h:h+8)+G*8
// C/D layout (16x16 f32): lane l, vgpr v -> (M=v+8*G, N=l%16)
// ---------------------------------------------------------------------------
template<int FIN_X, int G>
__device__ __forceinline__ v16h build_edge_row(const float* __restrict__ x,
                                               const float* __restrict__ pos,
                                               int src, float px0, float px1, float px2)
{
    constexpr int FIN = FIN_X + 3;
    v16h a;
    #pragma unroll
    for (int h = 0; h < 16; ++h) {
        const int k = ((h < 8) ? h : (h + 8)) + G * 8;   // compile-time
        float val;
        if (k < FIN_X) {                                 // folds at compile time
            val = x[src * FIN_X + k];                    // contiguous -> b128 merges
        } else if (k < FIN) {
            const int c = k - FIN_X;
            const float pp = (c == 0) ? px0 : ((c == 1) ? px1 : px2);
            val = pos[src * 3 + c] - pp;
        } else {
            val = 0.0f;                                  // K padding
        }
        a[h] = (_Float16)val;
    }
    return a;
}

template<int FIN, int FO, int G>
__device__ __forceinline__ v16h build_b_col(const float* __restrict__ w, int n)
{
    const int nc = (n < FO) ? n : (FO - 1);              // clamp: load stays unconditional
    v16h b;
    #pragma unroll
    for (int h = 0; h < 16; ++h) {
        const int k = ((h < 8) ? h : (h + 8)) + G * 8;   // compile-time
        float val = (k < FIN) ? w[k * FO + nc] : 0.0f;   // predicate folds
        if (n >= FO) val = 0.0f;                         // v_cndmask, no branch
        b[h] = (_Float16)val;
    }
    return b;
}

template<int FI, int G>
__device__ __forceinline__ v16h build_fc_row(const float* __restrict__ feat, int row)
{
    v16h a;
    #pragma unroll
    for (int h = 0; h < 16; ++h) {
        const int k = ((h < 8) ? h : (h + 8)) + G * 8;   // compile-time
        float val = (k < FI) ? feat[row * FI + k] : 0.0f;
        a[h] = (_Float16)val;
    }
    return a;
}

// ---------------------------------------------------------------------------
// Edge kernel: agg[p, fo] = max_{j<64} celu( [x[src_j], pos[src_j]-pos[p]] @ wa + ba )
// One wave per point; 4 WMMA M-tiles of 16 edges.
// ---------------------------------------------------------------------------
template<int FIN_X, int FO>
__global__ __launch_bounds__(256)
void edge_conv_wmma(const float* __restrict__ x,        // (N, FIN_X); == pos for layer 1
                    const float* __restrict__ pos,      // (N, 3)
                    const int*   __restrict__ in_index, // (N*64)
                    const float* __restrict__ wa,       // (FIN_X+3, FO)
                    const float* __restrict__ ba,       // (FO)
                    float*       __restrict__ agg)      // (N, FO)
{
    constexpr int FIN = FIN_X + 3;
    constexpr int NT  = (FO + 15) / 16;

    const int lane = threadIdx.x & 31;
    const int wave = threadIdx.x >> 5;
    const int p    = blockIdx.x * 8 + wave;
    const int g    = lane >> 4;
    const int col  = lane & 15;

    // Weights/bias in WMMA B/C layout (one divergent branch, vectorizable loads)
    v16h b[NT];
    v8f  cbias[NT];
    if (g == 0) {
        #pragma unroll
        for (int nt = 0; nt < NT; ++nt) b[nt] = build_b_col<FIN, FO, 0>(wa, col + nt * 16);
    } else {
        #pragma unroll
        for (int nt = 0; nt < NT; ++nt) b[nt] = build_b_col<FIN, FO, 1>(wa, col + nt * 16);
    }
    #pragma unroll
    for (int nt = 0; nt < NT; ++nt) {
        const int n = col + nt * 16;
        float bias = ba[(n < FO) ? n : (FO - 1)];
        if (n >= FO) bias = 0.0f;
        #pragma unroll
        for (int v = 0; v < 8; ++v) cbias[nt][v] = bias;
    }

    const float px0 = pos[p * 3 + 0];
    const float px1 = pos[p * 3 + 1];
    const float px2 = pos[p * 3 + 2];

    // Hoist all 4 tiles' neighbor indices so the index fetches overlap
    int srcs[4];
    #pragma unroll
    for (int t = 0; t < 4; ++t)
        srcs[t] = in_index[p * KNBR + t * 16 + (lane & 15)];

    float rmax[NT][8];
    #pragma unroll
    for (int nt = 0; nt < NT; ++nt)
        #pragma unroll
        for (int v = 0; v < 8; ++v) rmax[nt][v] = -INFINITY;

    #pragma unroll
    for (int t = 0; t < 4; ++t) {
        v16h a;
        if (g == 0) a = build_edge_row<FIN_X, 0>(x, pos, srcs[t], px0, px1, px2);
        else        a = build_edge_row<FIN_X, 1>(x, pos, srcs[t], px0, px1, px2);

        #pragma unroll
        for (int nt = 0; nt < NT; ++nt) {
            v8f c = __builtin_amdgcn_wmma_f32_16x16x32_f16(
                false, a, false, b[nt], (short)0, cbias[nt], false, false);
            #pragma unroll
            for (int v = 0; v < 8; ++v)
                rmax[nt][v] = fmaxf(rmax[nt][v], celu1(c[v]));
        }
    }

    // Segmented max: 8 rows per lane, then combine the two row-groups (wave32 xor-16)
    #pragma unroll
    for (int nt = 0; nt < NT; ++nt) {
        float m = rmax[nt][0];
        #pragma unroll
        for (int v = 1; v < 8; ++v) m = fmaxf(m, rmax[nt][v]);
        m = fmaxf(m, __shfl_xor(m, 16, 32));
        const int n = col + nt * 16;
        if (g == 0 && n < FO) agg[p * FO + n] = m;
    }
}

// ---------------------------------------------------------------------------
// Point kernel: out[p] = celu( agg[p] @ wb + bb ).  One wave per 16 points.
// ---------------------------------------------------------------------------
template<int FI, int FO>
__global__ __launch_bounds__(256)
void point_fc_wmma(const float* __restrict__ agg, // (N, FI)
                   const float* __restrict__ wb,  // (FI, FO)
                   const float* __restrict__ bb,  // (FO)
                   float*       __restrict__ out) // (N, FO)
{
    constexpr int NT = (FO + 15) / 16;
    const int lane = threadIdx.x & 31;
    const int wave = threadIdx.x >> 5;
    const int base = (blockIdx.x * 8 + wave) * 16;
    const int g    = lane >> 4;
    const int col  = lane & 15;

    v16h b[NT];
    v8f  cbias[NT];
    v16h a;
    if (g == 0) {
        #pragma unroll
        for (int nt = 0; nt < NT; ++nt) b[nt] = build_b_col<FI, FO, 0>(wb, col + nt * 16);
        a = build_fc_row<FI, 0>(agg, base + col);
    } else {
        #pragma unroll
        for (int nt = 0; nt < NT; ++nt) b[nt] = build_b_col<FI, FO, 1>(wb, col + nt * 16);
        a = build_fc_row<FI, 1>(agg, base + col);
    }
    #pragma unroll
    for (int nt = 0; nt < NT; ++nt) {
        const int n = col + nt * 16;
        float bias = bb[(n < FO) ? n : (FO - 1)];
        if (n >= FO) bias = 0.0f;
        #pragma unroll
        for (int v = 0; v < 8; ++v) cbias[nt][v] = bias;
    }

    #pragma unroll
    for (int nt = 0; nt < NT; ++nt) {
        v8f c = __builtin_amdgcn_wmma_f32_16x16x32_f16(
            false, a, false, b[nt], (short)0, cbias[nt], false, false);
        #pragma unroll
        for (int v = 0; v < 8; ++v) {
            const int pt = base + v + 8 * g;
            const int n  = col + nt * 16;
            if (n < FO) out[pt * FO + n] = celu1(c[v]);
        }
    }
}

// d_out = concat(pos[N*3], out[N*32], batch zeros). Copy pos, zero the tail;
// the middle N*32 region is written by the layer-3 point kernel.
__global__ void init_out_kernel(const float* __restrict__ pos,
                                float* __restrict__ out, int out_size) {
    const int i = blockIdx.x * blockDim.x + threadIdx.x;
    if (i >= out_size) return;
    if (i < 3 * NPTS)        out[i] = pos[i];
    else if (i >= 35 * NPTS) out[i] = 0.0f;
}

extern "C" void kernel_launch(void* const* d_in, const int* in_sizes, int n_in,
                              void* d_out, int out_size, void* d_ws, size_t ws_size,
                              hipStream_t stream) {
    (void)in_sizes; (void)n_in; (void)ws_size;
    const float* pos      = (const float*)d_in[0];
    // d_in[1]=rgb (unused), d_in[2]=batch (zeros), d_in[3]=out_index (implicit: e/64)
    const int*   in_index = (const int*)  d_in[4];
    const float* w1a = (const float*)d_in[5];
    const float* b1a = (const float*)d_in[6];
    const float* w1b = (const float*)d_in[7];
    const float* b1b = (const float*)d_in[8];
    const float* w2a = (const float*)d_in[9];
    const float* b2a = (const float*)d_in[10];
    const float* w2b = (const float*)d_in[11];
    const float* b2b = (const float*)d_in[12];
    const float* w3a = (const float*)d_in[13];
    const float* b3a = (const float*)d_in[14];
    const float* w3b = (const float*)d_in[15];
    const float* b3b = (const float*)d_in[16];

    float* ws   = (float*)d_ws;
    float* agg  = ws;                       // N*32 f32, reused by all 3 layers
    float* x1   = ws + 32 * NPTS;           // N*8
    float* x2   = ws + 40 * NPTS;           // N*16
    float* outF = (float*)d_out + 3 * NPTS; // N*32 region of the output tuple

    init_out_kernel<<<(out_size + 255) / 256, 256, 0, stream>>>(pos, (float*)d_out, out_size);

    const dim3 blk(256);
    const dim3 egrid(NPTS / 8);    // 1 wave per point, 8 waves per block
    const dim3 pgrid(NPTS / 128);  // 16 points per wave, 8 waves per block

    // Layer 1: msg = [pos[src], dpos] (6) -> 8 ; fc 8 -> 8
    edge_conv_wmma<3, 8 ><<<egrid, blk, 0, stream>>>(pos, pos, in_index, w1a, b1a, agg);
    point_fc_wmma <8, 8 ><<<pgrid, blk, 0, stream>>>(agg, w1b, b1b, x1);
    // Layer 2: msg = [x1[src], dpos] (11) -> 16 ; fc 16 -> 16
    edge_conv_wmma<8, 16><<<egrid, blk, 0, stream>>>(x1, pos, in_index, w2a, b2a, agg);
    point_fc_wmma <16,16><<<pgrid, blk, 0, stream>>>(agg, w2b, b2b, x2);
    // Layer 3: msg = [x2[src], dpos] (19) -> 32 ; fc 32 -> 32, write into d_out
    edge_conv_wmma<16,32><<<egrid, blk, 0, stream>>>(x2, pos, in_index, w3a, b3a, agg);
    point_fc_wmma <32,32><<<pgrid, blk, 0, stream>>>(agg, w3b, b3b, outF);
}
